// _AAConv2d_44495861186604
// MI455X (gfx1250) — matrix operand order repaired and checked by reading, not compiled, not run on hardware
//
#include <hip/hip_runtime.h>
#include <hip/hip_bf16.h>

typedef __attribute__((ext_vector_type(16))) __bf16 v16bf;
typedef __attribute__((ext_vector_type(8)))  __bf16 v8bf;
typedef __attribute__((ext_vector_type(8)))  float  v8f;

#ifndef __has_builtin
#define __has_builtin(x) 0
#endif
#if __has_builtin(__builtin_amdgcn_tensor_load_to_lds) && \
    __has_builtin(__builtin_amdgcn_s_wait_tensorcnt)
#define HAVE_TDM 1
#else
#define HAVE_TDM 0
#endif

namespace {

constexpr int Bn = 16, Cn = 256, Hn = 64, Wn = 64, HWn = 4096, Pn = 1024;
constexpr int HEADSn = 8, DHn = 32, KCn = 2304;  // 9 taps * 256 ch

union U16 { v16bf v; v8bf h[2]; };

__device__ inline v8bf zero8() {
  union { v8bf v; int i[4]; } z; z.i[0] = z.i[1] = z.i[2] = z.i[3] = 0; return z.v;
}

// A-matrix bf16 16x32 fragment (row-major source, stride ld):
// lane L<16: row m0+L, halves[0..7]=K k0..k0+7, halves[8..15]=K k0+16..k0+23
// lane L>=16: row m0+L-16, K base +8 / +24  (ISA 7.12.2 16-bit A layout)
__device__ inline v16bf load_a_frag(const __bf16* __restrict__ base, int m0, int ld,
                                    int k0, int lane) {
  int r  = m0 + (lane & 15);
  int kk = k0 + ((lane >> 4) << 3);
  U16 u;
  const __bf16* p = base + (long)r * ld + kk;
  u.h[0] = *(const v8bf*)(p);
  u.h[1] = *(const v8bf*)(p + 16);
  return u.v;
}

// B-matrix bf16 32x16 fragment from BT stored [N][K] row-major (stride ld):
// lane L<16: col n0+L, K k0..k0+7,k0+16..k0+23 ; lane L>=16: col n0+L-16, K base +8 / +24
__device__ inline v16bf load_b_frag(const __bf16* __restrict__ bt, int n0, int ld,
                                    int k0, int lane) {
  int n  = n0 + (lane & 15);
  int kk = k0 + ((lane >> 4) << 3);
  U16 u;
  const __bf16* p = bt + (long)n * ld + kk;
  u.h[0] = *(const v8bf*)(p);
  u.h[1] = *(const v8bf*)(p + 16);
  return u.v;
}

// B fragment from an LDS-staged 64(N) x 32(K) tile (row stride 32 elems)
__device__ inline v16bf load_b_frag_lds(const __bf16* sm, int n0, int lane) {
  int n  = n0 + (lane & 15);
  int kk = (lane >> 4) << 3;
  const __bf16* p = sm + n * 32 + kk;
  U16 u;
  u.h[0] = *(const v8bf*)(p);
  u.h[1] = *(const v8bf*)(p + 16);
  return u.v;
}

__device__ inline v8f wmma_bf16(v16bf a, v16bf b, v8f c) {
  return __builtin_amdgcn_wmma_f32_16x16x32_bf16(false, a, false, b, (short)0, c,
                                                 false, false);
}

// Generic LDS pointer -> LDS byte offset (ISA: LDS aperture; offset is the low 32 address bits)
__device__ inline unsigned lds_off(const void* p) {
  return (unsigned)(unsigned long long)p;
}

#if HAVE_TDM
typedef unsigned int __attribute__((ext_vector_type(4))) u32x4;
typedef int __attribute__((ext_vector_type(8)))          i32x8;
typedef int __attribute__((ext_vector_type(4)))          i32x4;

// Issue a TDM 2-D tile load: 32 elems (64B) x 64 rows of bf16 from BT[N][KCn]
// into LDS at ldsOff. Tracked by TENSORcnt.
__device__ inline void tdm_issue_b_tile(const __bf16* gsrc, unsigned ldsOff) {
  unsigned long long ga = (unsigned long long)(const void*)gsrc;
  u32x4 g0;
  g0[0] = 1u;                                                  // count=1, user D#
  g0[1] = ldsOff;                                              // lds_addr (bytes)
  g0[2] = (unsigned)(ga & 0xffffffffu);                        // global_addr[31:0]
  g0[3] = (unsigned)((ga >> 32) & 0x01ffffffu) | (2u << 30);   // addr[56:32] | type=2
  i32x8 g1;
  g1[0] = 1 << 16;                                             // data_size=1 -> 2 bytes
  g1[1] = (int)(((unsigned)KCn & 0xffffu) << 16);              // tensor_dim0 lo16
  g1[2] = (int)((((unsigned)KCn >> 16) & 0xffffu) |            // tensor_dim0 hi16
                (((unsigned)Cn & 0xffffu) << 16));             // tensor_dim1 lo16
  g1[3] = (int)((((unsigned)Cn >> 16) & 0xffffu) | (32u << 16)); // dim1 hi | tile_dim0=32
  g1[4] = 64;                                                  // tile_dim1=64, tile_dim2=0
  g1[5] = KCn;                                                 // tensor_dim0_stride lo32
  g1[6] = 0;                                                   // stride0 hi | stride1 lo
  g1[7] = 0;                                                   // stride1 hi
  i32x4 z4; z4[0] = z4[1] = z4[2] = z4[3] = 0;
  i32x8 z8;
#pragma unroll
  for (int i = 0; i < 8; i++) z8[i] = 0;
  // 6-arg form (clang-23 / therock-10.0): groups 2/3 unused for 2-D tiles
  __builtin_amdgcn_tensor_load_to_lds(g0, g1, z4, z4, z8, 0);
}
#endif

// ---------------- layout / repack kernels ----------------

__global__ void k_x_to_nhwc(const float* __restrict__ x, __bf16* __restrict__ xb) {
  __shared__ float tile[32][33];
  int b = blockIdx.z;
  int p0 = blockIdx.x * 32, c0 = blockIdx.y * 32;
  int tx = threadIdx.x, ty = threadIdx.y;
  tile[ty][tx] = x[((long)b * Cn + (c0 + ty)) * HWn + (p0 + tx)];
  __syncthreads();
  xb[((long)b * HWn + (p0 + ty)) * Cn + (c0 + tx)] = (__bf16)tile[tx][ty];
}

__global__ void k_v_transpose(const __bf16* __restrict__ V, __bf16* __restrict__ VT) {
  __shared__ __bf16 tile[32][33];
  int b = blockIdx.z;
  int p0 = blockIdx.x * 32, n0 = blockIdx.y * 32;
  int tx = threadIdx.x, ty = threadIdx.y;
  tile[ty][tx] = V[((long)b * Pn + (p0 + ty)) * Cn + (n0 + tx)];
  __syncthreads();
  VT[((long)b * Cn + (n0 + ty)) * Pn + (p0 + tx)] = tile[tx][ty];
}

__global__ void k_pack_fr(const float* __restrict__ s, __bf16* __restrict__ d) {
  int t = blockIdx.x * blockDim.x + threadIdx.x;
  if (t < Cn * Cn) d[t] = (__bf16)s[t];
}

// wq/wk/wv [in][out] -> BT[n][k] = w[k][(n%32)*8 + n/32]  (head-contiguous cols)
__global__ void k_pack_qkv(const float* __restrict__ s, __bf16* __restrict__ d) {
  int t = blockIdx.x * blockDim.x + threadIdx.x;
  if (t >= Cn * Cn) return;
  int n = t >> 8, k = t & 255;
  d[t] = (__bf16)s[k * Cn + ((n & 31) << 3) + (n >> 5)];
}

// wo [in][out] -> BT[n][k] = wo[(k%32)*8 + k/32][n]
__global__ void k_pack_wo(const float* __restrict__ s, __bf16* __restrict__ d) {
  int t = blockIdx.x * blockDim.x + threadIdx.x;
  if (t >= Cn * Cn) return;
  int n = t >> 8, k = t & 255;
  d[t] = (__bf16)s[(((k & 31) << 3) + (k >> 5)) * Cn + n];
}

// conv_w [O][I][3][3] -> [O][tap][I] bf16
__global__ void k_pack_conv(const float* __restrict__ s, __bf16* __restrict__ d) {
  int t = blockIdx.x * blockDim.x + threadIdx.x;
  if (t >= Cn * KCn) return;
  int o = t / KCn, rem = t % KCn, tap = rem >> 8, i = rem & 255;
  d[t] = (__bf16)s[(o * Cn + i) * 9 + tap];
}

// ---------------- GEMM: Y[b] = A[b] * BT^T (+bias) ----------------
template <int OUTMODE>  // 0 = f32 out, 1 = bf16 out
__global__ void k_gemm(const __bf16* __restrict__ A, long aBatch,
                       const __bf16* __restrict__ BT,
                       void* __restrict__ Y, long yBatch,
                       const float* __restrict__ bias,
                       int lda, int ldy, int K) {
  int lane = threadIdx.x & 31, wave = threadIdx.x >> 5;
  int m0 = blockIdx.x * 128 + wave * 16;
  int n0 = blockIdx.y * 64;
  const __bf16* Ab = A + (long)blockIdx.z * aBatch;
  v8f acc[4];
#pragma unroll
  for (int t = 0; t < 4; t++)
#pragma unroll
    for (int r = 0; r < 8; r++) acc[t][r] = 0.f;

  for (int k0 = 0; k0 < K; k0 += 32) {
    v16bf a = load_a_frag(Ab, m0, lda, k0, lane);
#pragma unroll
    for (int t = 0; t < 4; t++) {
      v16bf bf = load_b_frag(BT, n0 + t * 16, K, k0, lane);
      acc[t] = wmma_bf16(a, bf, acc[t]);
    }
  }
  int nc = lane & 15, rb = (lane >> 4) << 3;
#pragma unroll
  for (int t = 0; t < 4; t++) {
    int col = n0 + t * 16 + nc;
    float bv = bias ? bias[col] : 0.f;
#pragma unroll
    for (int r = 0; r < 8; r++) {
      long off = (long)blockIdx.z * yBatch + (long)(m0 + r + rb) * ldy + col;
      float v = acc[t][r] + bv;
      if (OUTMODE == 0) ((float*)Y)[off] = v;
      else              ((__bf16*)Y)[off] = (__bf16)v;
    }
  }
}

// ------- 3x3 conv, implicit GEMM, TDM-staged weight tiles in LDS -------
__global__ void k_conv3x3(const __bf16* __restrict__ xb, const __bf16* __restrict__ WT,
                          const float* __restrict__ bias, float* __restrict__ out) {
  __shared__ __bf16 bstage[2][64 * 32];  // double-buffered 64N x 32K weight tile
  int lane = threadIdx.x & 31, wave = threadIdx.x >> 5;
  int m0 = blockIdx.x * 128 + wave * 16;
  int n0 = blockIdx.y * 64;
  const __bf16* Xb = xb + (long)blockIdx.z * HWn * Cn;
  const __bf16* Wb = WT + (long)n0 * KCn;  // tile row 0; K offset = 32*it (linear)
  int p = m0 + (lane & 15);
  int h = p >> 6, w = p & 63;
  int klane = (lane >> 4) << 3;
  v8f acc[4];
#pragma unroll
  for (int t = 0; t < 4; t++)
#pragma unroll
    for (int r = 0; r < 8; r++) acc[t][r] = 0.f;

#if HAVE_TDM
  if (wave == 0) {
    tdm_issue_b_tile(Wb, lds_off(&bstage[0][0]));
    __builtin_amdgcn_s_wait_tensorcnt(0);
  }
  __syncthreads();
#endif

#pragma unroll
  for (int tap = 0; tap < 9; ++tap) {
    const int dh = tap / 3 - 1, dw = tap % 3 - 1;  // constants after unroll
    const int hs = h + dh, ws = w + dw;
    const bool inb = ((unsigned)hs < (unsigned)Hn) && ((unsigned)ws < (unsigned)Wn);
    const __bf16* arow = Xb + ((long)(hs * Wn + ws)) * Cn + klane;
    for (int j = 0; j < 8; ++j) {
      const int it = tap * 8 + j;
      const int buf = it & 1;
#if HAVE_TDM
      // prefetch next weight tile into the other buffer while computing
      if (it + 1 < 72 && wave == 0)
        tdm_issue_b_tile(Wb + (it + 1) * 32, lds_off(&bstage[buf ^ 1][0]));
#else
      // cooperative staged copy: 256 threads x 8 elems = 64x32 tile
      {
        int t = threadIdx.x;
        int nr = t >> 2, k8 = (t & 3) << 3;
        *(v8bf*)&bstage[buf][nr * 32 + k8] =
            *(const v8bf*)(Wb + (long)nr * KCn + it * 32 + k8);
      }
      __syncthreads();
#endif
      U16 u;
      if (inb) {
        const __bf16* src = arow + j * 32;
        u.h[0] = *(const v8bf*)(src);
        u.h[1] = *(const v8bf*)(src + 16);
      } else {
        u.h[0] = zero8(); u.h[1] = zero8();
      }
#pragma unroll
      for (int t4 = 0; t4 < 4; t4++) {
        v16bf bf = load_b_frag_lds(&bstage[buf][0], t4 * 16, lane);
        acc[t4] = wmma_bf16(u.v, bf, acc[t4]);
      }
#if HAVE_TDM
      if (wave == 0) __builtin_amdgcn_s_wait_tensorcnt(0);
#endif
      __syncthreads();
    }
  }

  int nc = lane & 15, rb = (lane >> 4) << 3;
#pragma unroll
  for (int t = 0; t < 4; t++) {
    int col = n0 + t * 16 + nc;
    float bv = bias[col];
#pragma unroll
    for (int r = 0; r < 8; r++)
      out[(long)blockIdx.z * HWn * Cn + (long)(m0 + r + rb) * Cn + col] =
          acc[t][r] + bv;
  }
}

// ---------------- maxpool 2x2 with argmax (first-max tie-break) ----------------
__global__ void k_pool(const __bf16* __restrict__ y1, __bf16* __restrict__ pooled,
                       unsigned char* __restrict__ pidx, int total) {
  int t = blockIdx.x * blockDim.x + threadIdx.x;
  if (t >= total) return;
  int c = t & 255;
  int pt = (t >> 8) & (Pn - 1);
  int b = t >> 18;
  int ph = pt >> 5, pw = pt & 31;
  float best = -__builtin_inff(); int bi = 0;
#pragma unroll
  for (int j = 0; j < 4; j++) {
    int hh = 2 * ph + (j >> 1), ww = 2 * pw + (j & 1);
    float v = (float)y1[((long)b * HWn + hh * Wn + ww) * Cn + c];
    if (v > best) { best = v; bi = j; }
  }
  pooled[t] = (__bf16)best;
  pidx[t] = (unsigned char)bi;
}

// ---------------- flash attention: 1 wave per (b, head, 16 queries) ----------------
__global__ void k_attn(const __bf16* __restrict__ Qm, const __bf16* __restrict__ Km,
                       const __bf16* __restrict__ VTm, __bf16* __restrict__ Om) {
  __shared__ float  Sf[16][32];
  __shared__ __bf16 Pl[16][32];
  __shared__ float  mrow[16], lrow[16], crow[16];
  int lane = threadIdx.x;
  int qb = blockIdx.x, head = blockIdx.y, b = blockIdx.z;
  const __bf16* Qb_ = Qm + (long)b * Pn * Cn + head * DHn;
  const __bf16* Kb_ = Km + (long)b * Pn * Cn + head * DHn;
  const __bf16* Vt_ = VTm + ((long)b * Cn + head * DHn) * Pn;

  v16bf qf = load_a_frag(Qb_, qb * 16, Cn, 0, lane);
  const float scale = 0.17677669529663687f;  // 1/sqrt(32)
#pragma unroll
  for (int i = 0; i < 16; i++) qf[i] = (__bf16)((float)qf[i] * scale);

  v8f o0, o1;
#pragma unroll
  for (int r = 0; r < 8; r++) { o0[r] = 0.f; o1[r] = 0.f; }
  if (lane < 16) { mrow[lane] = -__builtin_inff(); lrow[lane] = 0.f; }
  __syncthreads();

  int nc = lane & 15, rb = (lane >> 4) << 3;
  for (int kb = 0; kb < Pn; kb += 32) {
    v16bf kf0 = load_b_frag(Kb_, kb,      Cn, 0, lane);
    v16bf kf1 = load_b_frag(Kb_, kb + 16, Cn, 0, lane);
    v8f z;
#pragma unroll
    for (int r = 0; r < 8; r++) z[r] = 0.f;
    v8f s0 = wmma_bf16(qf, kf0, z);
    v8f s1 = wmma_bf16(qf, kf1, z);
#pragma unroll
    for (int r = 0; r < 8; r++) { Sf[r + rb][nc] = s0[r]; Sf[r + rb][nc + 16] = s1[r]; }
    __syncthreads();
    if (lane < 16) {
      float mOld = mrow[lane], mb = mOld;
      for (int j = 0; j < 32; j++) mb = fmaxf(mb, Sf[lane][j]);
      float corr = __expf(mOld - mb);
      float sum = 0.f;
      for (int j = 0; j < 32; j++) {
        float e = __expf(Sf[lane][j] - mb);
        Pl[lane][j] = (__bf16)e;
        sum += e;
      }
      lrow[lane] = lrow[lane] * corr + sum;
      mrow[lane] = mb;
      crow[lane] = corr;
    }
    __syncthreads();
#pragma unroll
    for (int r = 0; r < 8; r++) { float cf = crow[r + rb]; o0[r] *= cf; o1[r] *= cf; }
    v16bf pf;
    {
      U16 u;
      const __bf16* pp = &Pl[lane & 15][(lane >> 4) << 3];
      u.h[0] = *(const v8bf*)(pp);
      u.h[1] = *(const v8bf*)(pp + 16);
      pf = u.v;
    }
    v16bf vf0 = load_b_frag(Vt_, 0,  Pn, kb, lane);
    v16bf vf1 = load_b_frag(Vt_, 16, Pn, kb, lane);
    o0 = wmma_bf16(pf, vf0, o0);
    o1 = wmma_bf16(pf, vf1, o1);
    __syncthreads();
  }
  if (lane < 16) crow[lane] = 1.f / lrow[lane];
  __syncthreads();
#pragma unroll
  for (int r = 0; r < 8; r++) {
    int q = qb * 16 + r + rb;
    float inv = crow[r + rb];
    Om[((long)b * Pn + q) * Cn + head * DHn + nc]      = (__bf16)(o0[r] * inv);
    Om[((long)b * Pn + q) * Cn + head * DHn + nc + 16] = (__bf16)(o1[r] * inv);
  }
}

// ---------------- unpool + tanh(gamma)*attn + conv add, NCHW f32 out ----------------
__global__ void k_fuse(const float* __restrict__ conv_out, const float* __restrict__ attnP,
                       const unsigned char* __restrict__ pidx,
                       const float* __restrict__ gamma,
                       float* __restrict__ out, int total) {
  int t = blockIdx.x * blockDim.x + threadIdx.x;
  if (t >= total) return;
  int w = t & 63, h = (t >> 6) & 63, c = (t >> 12) & 255, b = t >> 20;
  float cv = conv_out[((long)b * HWn + (h * Wn + w)) * Cn + c];
  int pt = (h >> 1) * 32 + (w >> 1);
  int quad = ((h & 1) << 1) | (w & 1);
  long pi = ((long)b * Pn + pt) * Cn + c;
  float av = (pidx[pi] == (unsigned char)quad) ? attnP[pi] : 0.f;
  out[t] = av * tanhf(gamma[0]) + cv;
}

}  // namespace

extern "C" void kernel_launch(void* const* d_in, const int* in_sizes, int n_in,
                              void* d_out, int out_size, void* d_ws, size_t ws_size,
                              hipStream_t stream) {
  const float* x      = (const float*)d_in[0];
  const float* conv_w = (const float*)d_in[1];
  const float* conv_b = (const float*)d_in[2];
  const float* fr_w   = (const float*)d_in[3];
  const float* fr_b   = (const float*)d_in[4];
  const float* wq     = (const float*)d_in[5];
  const float* wk     = (const float*)d_in[6];
  const float* wv     = (const float*)d_in[7];
  const float* wo     = (const float*)d_in[8];
  const float* gamma  = (const float*)d_in[9];

  char* wp = (char*)d_ws;
  auto alloc = [&](size_t bytes) -> void* {
    void* p = (void*)wp;
    wp += (bytes + 255) & ~(size_t)255;
    return p;
  };
  __bf16* xb      = (__bf16*)alloc((size_t)Bn * HWn * Cn * 2);
  __bf16* y1      = (__bf16*)alloc((size_t)Bn * HWn * Cn * 2);
  __bf16* pooledT = (__bf16*)alloc((size_t)Bn * Pn * Cn * 2);
  unsigned char* pidx = (unsigned char*)alloc((size_t)Bn * Pn * Cn);
  __bf16* Qb      = (__bf16*)alloc((size_t)Bn * Pn * Cn * 2);
  __bf16* Kb      = (__bf16*)alloc((size_t)Bn * Pn * Cn * 2);
  __bf16* Vb      = (__bf16*)alloc((size_t)Bn * Pn * Cn * 2);
  __bf16* VTb     = (__bf16*)alloc((size_t)Bn * Pn * Cn * 2);
  __bf16* attnO   = (__bf16*)alloc((size_t)Bn * Pn * Cn * 2);
  float*  attnP   = (float*)alloc((size_t)Bn * Pn * Cn * 4);
  float*  convO   = (float*)alloc((size_t)Bn * HWn * Cn * 4);
  __bf16* frT     = (__bf16*)alloc((size_t)Cn * Cn * 2);
  __bf16* wqT     = (__bf16*)alloc((size_t)Cn * Cn * 2);
  __bf16* wkT     = (__bf16*)alloc((size_t)Cn * Cn * 2);
  __bf16* wvT     = (__bf16*)alloc((size_t)Cn * Cn * 2);
  __bf16* woT     = (__bf16*)alloc((size_t)Cn * Cn * 2);
  __bf16* convT   = (__bf16*)alloc((size_t)Cn * KCn * 2);
  (void)in_sizes; (void)n_in; (void)out_size; (void)ws_size;

  dim3 t32x32(32, 32);

  k_x_to_nhwc<<<dim3(HWn / 32, Cn / 32, Bn), t32x32, 0, stream>>>(x, xb);
  k_pack_fr  <<<(Cn * Cn) / 256, 256, 0, stream>>>(fr_w, frT);
  k_pack_qkv <<<(Cn * Cn) / 256, 256, 0, stream>>>(wq, wqT);
  k_pack_qkv <<<(Cn * Cn) / 256, 256, 0, stream>>>(wk, wkT);
  k_pack_qkv <<<(Cn * Cn) / 256, 256, 0, stream>>>(wv, wvT);
  k_pack_wo  <<<(Cn * Cn) / 256, 256, 0, stream>>>(wo, woT);
  k_pack_conv<<<(Cn * KCn) / 256, 256, 0, stream>>>(conv_w, convT);

  // 1x1 feature reduction: y1 = xb @ frT^T + fr_b (bf16 out)
  k_gemm<1><<<dim3(HWn / 128, Cn / 64, Bn), 256, 0, stream>>>(
      xb, (long)HWn * Cn, frT, (void*)y1, (long)HWn * Cn, fr_b, Cn, Cn, Cn);

  k_pool<<<(Bn * Pn * Cn) / 256, 256, 0, stream>>>(y1, pooledT, pidx, Bn * Pn * Cn);

  k_gemm<1><<<dim3(Pn / 128, Cn / 64, Bn), 256, 0, stream>>>(
      pooledT, (long)Pn * Cn, wqT, (void*)Qb, (long)Pn * Cn, nullptr, Cn, Cn, Cn);
  k_gemm<1><<<dim3(Pn / 128, Cn / 64, Bn), 256, 0, stream>>>(
      pooledT, (long)Pn * Cn, wkT, (void*)Kb, (long)Pn * Cn, nullptr, Cn, Cn, Cn);
  k_gemm<1><<<dim3(Pn / 128, Cn / 64, Bn), 256, 0, stream>>>(
      pooledT, (long)Pn * Cn, wvT, (void*)Vb, (long)Pn * Cn, nullptr, Cn, Cn, Cn);

  k_v_transpose<<<dim3(Pn / 32, Cn / 32, Bn), t32x32, 0, stream>>>(Vb, VTb);

  k_attn<<<dim3(Pn / 16, HEADSn, Bn), 32, 0, stream>>>(Qb, Kb, VTb, attnO);

  k_gemm<0><<<dim3(Pn / 128, Cn / 64, Bn), 256, 0, stream>>>(
      attnO, (long)Pn * Cn, woT, (void*)attnP, (long)Pn * Cn, nullptr, Cn, Cn, Cn);

  k_conv3x3<<<dim3(HWn / 128, Cn / 64, Bn), 256, 0, stream>>>(xb, convT, conv_b, convO);

  k_fuse<<<(Bn * Cn * HWn) / 256, 256, 0, stream>>>(convO, attnP, pidx, gamma,
                                                    (float*)d_out, Bn * Cn * HWn);
}
